// GINConv_9938554323125
// MI455X (gfx1250) — compile-verified, matrix operand order checked by reading, not run on hardware
//
#include <hip/hip_runtime.h>
#include <math.h>

typedef __attribute__((ext_vector_type(16))) __bf16          v16bf;
typedef __attribute__((ext_vector_type(8)))  float           v8f;
typedef __attribute__((ext_vector_type(8)))  unsigned short  u16x8;

union Frag {
  v16bf bf;
  unsigned short u[16];
  u16x8 h[2];     // h[0] = k..k+7, h[1] = k+16..k+23 (ISA 7.12.2 bf16 layout)
};

__device__ __forceinline__ unsigned short f2bf(float f) {
  union { float f; unsigned u; } c; c.f = f;
  unsigned u = c.u;
  u += 0x7FFFu + ((u >> 16) & 1u);   // round-to-nearest-even
  return (unsigned short)(u >> 16);
}

__device__ __forceinline__ unsigned pack2bf(float lo, float hi) {
  return (unsigned)f2bf(lo) | ((unsigned)f2bf(hi) << 16);
}

#define LDS_S 136   // padded row stride (272 B): optimal 2-way LDS banking
#define MT    32    // M-tile rows per block (2 WMMA tiles per wave)

// ---------------------------------------------------------------- h = x -----
__global__ __launch_bounds__(256) void gin_copy_x(const float4* __restrict__ x,
                                                  float4* __restrict__ h, int n4) {
  int i = blockIdx.x * 256 + threadIdx.x;
  if (i < n4) h[i] = x[i];
}

// ---------------------------------------------- h[dst] += x[src] per edge ---
__global__ __launch_bounds__(256) void gin_scatter(const float* __restrict__ x,
                                                   const int* __restrict__ ei,
                                                   float* __restrict__ h, int E) {
  int w    = (blockIdx.x * 256 + threadIdx.x) >> 5;
  int lane = threadIdx.x & 31;
  if (w >= E) return;
  int src = ei[w];       // edge_index[0][w]
  int dst = ei[E + w];   // edge_index[1][w]
  const float4 v = *(const float4*)(x + (size_t)src * 128 + lane * 4);
  float* p = h + (size_t)dst * 128 + lane * 4;
  atomicAdd(p + 0, v.x);
  atomicAdd(p + 1, v.y);
  atomicAdd(p + 2, v.z);
  atomicAdd(p + 3, v.w);
}

// --------------- out = elu(h@W1 + b1) @ W2 + b2, 32-row tile per block -----
__global__ __launch_bounds__(256) void gin_mlp(const float* __restrict__ W1,
                                               const float* __restrict__ b1,
                                               const float* __restrict__ W2,
                                               const float* __restrict__ b2,
                                               float* __restrict__ h,  // in/out = d_out
                                               int N) {
  __shared__ unsigned short sWt[128 * LDS_S];  // W^T, bf16, padded stride
  __shared__ unsigned short sA [MT  * LDS_S];  // input tile, bf16
  __shared__ unsigned short sH [MT  * LDS_S];  // hidden tile, bf16

  const int tid  = threadIdx.x;
  const int wave = tid >> 5;
  const int lane = tid & 31;
  const int m0   = blockIdx.x * MT;

  // Prefetch W2 (64 KB span) so it's cache-hot when we reload sWt.
  {
    int off = tid * 256;
    if (off < 128 * 128 * 4) __builtin_prefetch((const char*)W2 + off, 0, 1);
  }

  // sWt[n][k] = W1[k][n]; convert k-pairs and store packed b32 (LDS-contiguous)
  for (int i = tid; i < 64 * 128; i += 256) {
    int k2 = (i >> 7) << 1;
    int n  = i & 127;
    *(unsigned*)(sWt + n * LDS_S + k2) =
        pack2bf(W1[k2 * 128 + n], W1[(k2 + 1) * 128 + n]);
  }
  // A tile: float2 read, packed bf16 pair store
  for (int i = tid; i < MT * 64; i += 256) {
    int r  = i >> 6;
    int c2 = (i & 63) << 1;
    int gm = m0 + r;
    float2 v = (gm < N) ? *(const float2*)(h + (size_t)gm * 128 + c2)
                        : make_float2(0.0f, 0.0f);
    *(unsigned*)(sA + r * LDS_S + c2) = pack2bf(v.x, v.y);
  }
  __syncthreads();

  // ISA 7.12.2 bf16 fragment geometry (wave32)
  const int colL  = (wave << 4) + (lane & 15);  // this wave's N-column
  const int row   = lane & 15;                  // A row (M sub-tile 0)
  const int kb    = (lane >> 4) << 3;           // K sub-base per half-wave
  const int rbase = (lane >> 4) << 3;           // C/D row base per half-wave

  // ---- GEMM1: two 16x16 M-tiles per wave share each B fragment ----
  v8f acc0 = {}, acc1 = {};
#pragma unroll
  for (int kc = 0; kc < 128; kc += 32) {
    const int k0 = kc + kb;
    Frag a0, a1, b;
    b.h[0]  = *(const u16x8*)(sWt + colL * LDS_S + k0);
    b.h[1]  = *(const u16x8*)(sWt + colL * LDS_S + k0 + 16);
    a0.h[0] = *(const u16x8*)(sA + row * LDS_S + k0);
    a0.h[1] = *(const u16x8*)(sA + row * LDS_S + k0 + 16);
    a1.h[0] = *(const u16x8*)(sA + (16 + row) * LDS_S + k0);
    a1.h[1] = *(const u16x8*)(sA + (16 + row) * LDS_S + k0 + 16);
    acc0 = __builtin_amdgcn_wmma_f32_16x16x32_bf16(false, a0.bf, false, b.bf,
                                                   (short)0, acc0, false, false);
    acc1 = __builtin_amdgcn_wmma_f32_16x16x32_bf16(false, a1.bf, false, b.bf,
                                                   (short)0, acc1, false, false);
  }
  const float bias1 = b1[colL];
#pragma unroll
  for (int r = 0; r < 8; ++r) {
    float v0 = acc0[r] + bias1;
    float v1 = acc1[r] + bias1;
    v0 = v0 > 0.0f ? v0 : expm1f(v0);             // ELU (alpha = 1)
    v1 = v1 > 0.0f ? v1 : expm1f(v1);
    sH[(rbase + r) * LDS_S + colL]        = f2bf(v0);
    sH[(16 + rbase + r) * LDS_S + colL]   = f2bf(v1);
  }
  __syncthreads();

  // swap W2^T into the shared weight buffer
  for (int i = tid; i < 64 * 128; i += 256) {
    int k2 = (i >> 7) << 1;
    int n  = i & 127;
    *(unsigned*)(sWt + n * LDS_S + k2) =
        pack2bf(W2[k2 * 128 + n], W2[(k2 + 1) * 128 + n]);
  }
  __syncthreads();

  // ---- GEMM2 ----
  v8f acc2 = {}, acc3 = {};
#pragma unroll
  for (int kc = 0; kc < 128; kc += 32) {
    const int k0 = kc + kb;
    Frag a0, a1, b;
    b.h[0]  = *(const u16x8*)(sWt + colL * LDS_S + k0);
    b.h[1]  = *(const u16x8*)(sWt + colL * LDS_S + k0 + 16);
    a0.h[0] = *(const u16x8*)(sH + row * LDS_S + k0);
    a0.h[1] = *(const u16x8*)(sH + row * LDS_S + k0 + 16);
    a1.h[0] = *(const u16x8*)(sH + (16 + row) * LDS_S + k0);
    a1.h[1] = *(const u16x8*)(sH + (16 + row) * LDS_S + k0 + 16);
    acc2 = __builtin_amdgcn_wmma_f32_16x16x32_bf16(false, a0.bf, false, b.bf,
                                                   (short)0, acc2, false, false);
    acc3 = __builtin_amdgcn_wmma_f32_16x16x32_bf16(false, a1.bf, false, b.bf,
                                                   (short)0, acc3, false, false);
  }
  const float bias2 = b2[colL];
#pragma unroll
  for (int r = 0; r < 8; ++r) {
    int g0 = m0 + rbase + r;
    int g1 = m0 + 16 + rbase + r;
    if (g0 < N) h[(size_t)g0 * 128 + colL] = acc2[r] + bias2;
    if (g1 < N) h[(size_t)g1 * 128 + colL] = acc3[r] + bias2;
  }
}

// ---------------------------------------------------------------------------
extern "C" void kernel_launch(void* const* d_in, const int* in_sizes, int n_in,
                              void* d_out, int out_size, void* d_ws, size_t ws_size,
                              hipStream_t stream) {
  const float* x  = (const float*)d_in[0];
  const int*   ei = (const int*)d_in[1];
  const float* W1 = (const float*)d_in[2];
  const float* b1 = (const float*)d_in[3];
  const float* W2 = (const float*)d_in[4];
  const float* b2 = (const float*)d_in[5];
  float* out = (float*)d_out;

  const int ND = in_sizes[0];     // N * 128
  const int N  = ND / 128;        // 100000
  const int E  = in_sizes[1] / 2; // 1600000

  const int n4 = ND / 4;
  gin_copy_x<<<(n4 + 255) / 256, 256, 0, stream>>>((const float4*)x, (float4*)out, n4);
  gin_scatter<<<(E + 7) / 8, 256, 0, stream>>>(x, ei, out, E);
  gin_mlp<<<(N + MT - 1) / MT, 256, 0, stream>>>(W1, b1, W2, b2, out, N);
}
